// OnlineTopK_Fused_65936337928215
// MI455X (gfx1250) — matrix operand order repaired and checked by reading, not compiled, not run on hardware
//
#include <hip/hip_runtime.h>
#include <hip/hip_bf16.h>

// ---- problem hyperparameters (fixed by the reference) ----
#define B_    2
#define L_    8192
#define S_    128
#define H_    16
#define D_    128
#define TOPK  16
#define WIN   64
#define BLKSZ 64
// 1/sqrt(128)
#define SM_SCALE 0.08838834764831845f

#define WAVES_PER_BLOCK 4
#define SCORE_STRIDE 132   // 128 + 4 pad: 528B row stride -> conflict-free column reads

typedef __attribute__((ext_vector_type(16))) __bf16 v16bf;
typedef __attribute__((ext_vector_type(8)))  float  v8f;

__global__ __launch_bounds__(WAVES_PER_BLOCK * 32)
void OnlineTopK_Fused_kernel(const __hip_bfloat16* __restrict__ q,
                             const __hip_bfloat16* __restrict__ kmat,
                             float* __restrict__ out_vals,
                             int*   __restrict__ out_idx)
{
    // per-wave LDS regions (disjoint): 16x132 f32 scores + 16x16 topk val/idx
    __shared__ float s_scores[WAVES_PER_BLOCK][16 * SCORE_STRIDE];
    __shared__ float s_vals  [WAVES_PER_BLOCK][TOPK * 16];
    __shared__ int   s_idxs  [WAVES_PER_BLOCK][TOPK * 16];

    const int lane = threadIdx.x & 31;
    const int wib  = threadIdx.x >> 5;
    const int wave = blockIdx.x * WAVES_PER_BLOCK + wib;

    const int QT = L_ / 16;                    // 512 q-tiles per (b,h)
    const int qt = wave % QT;
    const int h  = (wave / QT) % H_;
    const int b  = wave / (QT * H_);

    const int qbase = qt * 16;
    const int m  = lane & 15;                  // row (A) / column (B,C) within tile
    const int hi = lane >> 4;                  // half-wave select

    const size_t rowStride = (size_t)H_ * D_;  // 2048 elements between rows/keys

    const __hip_bfloat16* qrow  = q    + (((size_t)b * L_ + qbase + m) * H_ + h) * D_;
    const __hip_bfloat16* kbase = kmat + (((size_t)b * S_) * H_ + h) * D_;

    // ---- load 4 A fragments (16x32 bf16 each, K = 0,32,64,96) ----
    // ISA 16-bit A layout: lanes 0-15 hold K-pairs {0..7,16..23}; lanes 16-31 {8..15,24..31}
    v16bf afrag[4];
    #pragma unroll
    for (int kk = 0; kk < 4; ++kk) {
        const int k0 = kk * 32;
        union { uint4 u[2]; v16bf v; } a;
        a.u[0] = *reinterpret_cast<const uint4*>(qrow + k0 + hi * 8);        // elems 0..7
        a.u[1] = *reinterpret_cast<const uint4*>(qrow + k0 + 16 + hi * 8);   // elems 8..15
        afrag[kk] = a.v;
    }

    float* scores = s_scores[wib];

    // ---- 8 s-tiles of 16 keys; 4 WMMAs (K=32) each; fp32 accum ----
    #pragma unroll
    for (int st = 0; st < 8; ++st) {
        const __hip_bfloat16* kcol = kbase + (size_t)(st * 16 + m) * rowStride;
        v8f acc = {0.f, 0.f, 0.f, 0.f, 0.f, 0.f, 0.f, 0.f};
        #pragma unroll
        for (int kk = 0; kk < 4; ++kk) {
            // ISA 16-bit B layout (matches A): lanes 0-15 hold K {0..7,16..23} of column n;
            // lanes 16-31 hold K {8..15,24..31}
            const int koff = kk * 32 + hi * 8;
            union { uint4 u[2]; v16bf v; } bf;
            bf.u[0] = *reinterpret_cast<const uint4*>(kcol + koff);
            bf.u[1] = *reinterpret_cast<const uint4*>(kcol + koff + 16);
            acc = __builtin_amdgcn_wmma_f32_16x16x32_bf16(
                false, afrag[kk], false, bf.v, (short)0, acc, false, false);
        }
        // C layout: lanes 0-15 hold rows 0..7 (vgpr r), lanes 16-31 rows 8..15; col = lane&15
        #pragma unroll
        for (int r = 0; r < 8; ++r) {
            const int row = hi * 8 + r;
            const int col = st * 16 + m;
            scores[row * SCORE_STRIDE + col] = acc[r] * SM_SCALE;
        }
    }

    // ---- per-row stable top-16 (lanes 0..15, one query row each) ----
    float* vals = s_vals[wib];
    int*   idxs = s_idxs[wib];
    if (lane < 16) {
        const int r  = lane;
        const int tq = qbase + r;

        #pragma unroll
        for (int j = 0; j < TOPK; ++j) {
            vals[j * 16 + r] = -__builtin_inff();
            idxs[j * 16 + r] = -1;
        }

        // causal prefix: allowed s < floor((tq - WIN + 1) / BLKSZ); negative -> none
        int limit = (tq - (WIN - 1)) >= 0 ? (tq - (WIN - 1)) / BLKSZ : 0;
        if (limit < 0) limit = 0;
        const int smax = limit < S_ ? limit : S_;

        for (int s = 0; s < smax; ++s) {
            const float v = scores[r * SCORE_STRIDE + s];
            if (v > vals[(TOPK - 1) * 16 + r]) {
                int p = TOPK - 1;
                while (p > 0 && v > vals[(p - 1) * 16 + r]) {
                    vals[p * 16 + r] = vals[(p - 1) * 16 + r];
                    idxs[p * 16 + r] = idxs[(p - 1) * 16 + r];
                    --p;
                }
                vals[p * 16 + r] = v;   // strict '>' insertion == stable ties (smaller idx first)
                idxs[p * 16 + r] = s;
            }
        }

        const size_t outOff = (((size_t)b * L_ + tq) * H_ + h) * TOPK;
        #pragma unroll
        for (int j = 0; j < TOPK; ++j) {
            out_vals[outOff + j] = vals[j * 16 + r];
            out_idx [outOff + j] = idxs[j * 16 + r];
        }
    }
}

extern "C" void kernel_launch(void* const* d_in, const int* in_sizes, int n_in,
                              void* d_out, int out_size, void* d_ws, size_t ws_size,
                              hipStream_t stream) {
    (void)in_sizes; (void)n_in; (void)d_ws; (void)ws_size; (void)out_size;

    const __hip_bfloat16* q = (const __hip_bfloat16*)d_in[0];
    const __hip_bfloat16* k = (const __hip_bfloat16*)d_in[1];

    const size_t nOut = (size_t)B_ * L_ * H_ * TOPK;   // 4,194,304
    float* out_vals = (float*)d_out;
    int*   out_idx  = (int*)((float*)d_out + nOut);    // tuple concat: vals then idx

    const int totalWaves = B_ * H_ * (L_ / 16);        // 16384
    const int blocks     = totalWaves / WAVES_PER_BLOCK;
    OnlineTopK_Fused_kernel<<<blocks, WAVES_PER_BLOCK * 32, 0, stream>>>(
        q, k, out_vals, out_idx);
}
